// WBLAttention_11570641895580
// MI455X (gfx1250) — matrix operand order repaired and checked by reading, not compiled, not run on hardware
//
#include <hip/hip_runtime.h>
#include <hip/hip_bf16.h>

// ---------------- types ----------------
typedef __attribute__((ext_vector_type(16))) __bf16 v16bf;
typedef __attribute__((ext_vector_type(4)))  __bf16 v4bf;
typedef __attribute__((ext_vector_type(8)))  float  v8f;
typedef __attribute__((ext_vector_type(4)))  unsigned int v4u;
typedef __attribute__((ext_vector_type(8)))  int    v8i;
typedef __attribute__((ext_vector_type(4)))  int    v4i;
typedef unsigned int u32;

union Frag16 { v16bf v; u32 u[8]; };

#define T_TOK   4096
#define HIDDEN  2048
#define N_HEADS 16
#define QK_NOPE 128
#define QK_ROPE 64
#define QK_DIM  192
#define V_DIM   128
#define Q_LORA  1536
#define KV_LORA 512
#define SCALING 0.07216878364870323f   // 192^-0.5
#define EPS     1e-6f

#define BM 128
#define BN 64
#define BK 32
#define LDA_PAD 8                       // bf16 elems -> 16B pad per 64B row
#define LDT (BK + LDA_PAD)              // 40 bf16 = 80B row pitch in LDS

// =====================================================================
// TDM 2D tile load (see CDNA5 ISA; 6-arg builtin)
// =====================================================================
__device__ __forceinline__
void tdm_load_2d(u32 lds_addr, const __bf16* gptr, u32 rows, u32 stride_elems)
{
    unsigned long long ga = (unsigned long long)(size_t)gptr;
    v4u g0;
    g0.x = 1u;                                            // count=1, user desc
    g0.y = lds_addr;                                      // LDS byte address
    g0.z = (u32)ga;                                       // global_addr[31:0]
    g0.w = ((u32)(ga >> 32) & 0x1FFFFFFu) | (2u << 30);   // addr[56:32] | type=2
    v8i g1;
    // data_size=2B(1)@16 | pad_enable@20 | pad_interval=3(16 dwords)@24:22
    // | pad_amount=3(4 dwords)@31:25
    g1[0] = (int)((1u << 16) | (1u << 20) | (3u << 22) | (3u << 25));
    g1[1] = (int)((u32)BK << 16);        // tensor_dim0[15:0] @ bits 63:48
    g1[2] = (int)(rows << 16);           // tensor_dim0 hi=0 | tensor_dim1 lo
    g1[3] = (int)((u32)BK << 16);        // tensor_dim1 hi=0 | tile_dim0=BK
    g1[4] = (int)rows;                   // tile_dim1=rows, tile_dim2=0
    g1[5] = (int)stride_elems;           // tensor_dim0_stride[31:0]
    g1[6] = 0;
    g1[7] = 0;
    v4i z4 = {0, 0, 0, 0};
    v8i z8 = {0, 0, 0, 0, 0, 0, 0, 0};
    __builtin_amdgcn_tensor_load_to_lds(g0, g1, z4, z4, z8, 0);
}

// =====================================================================
// GEMM: C[M,N] = A[M,K] @ BT[N,K]^T, bf16 in / f32 out, WMMA compute.
// Double-buffered LDS tiles fetched by the Tensor Data Mover.
// =====================================================================
template<int N, int K>
__global__ __launch_bounds__(128)
void gemm_tdm(const __bf16* __restrict__ A, const __bf16* __restrict__ BT,
              float* __restrict__ C)
{
    __shared__ __bf16 smem[2 * (BM + BN) * LDT];

    const int tid  = threadIdx.x;
    const int wave = tid >> 5;
    const int lane = tid & 31;
    const int half = lane >> 4;
    const int lr   = lane & 15;
    const int m0   = blockIdx.x * BM;
    const int n0   = blockIdx.y * BN;

    // flat addr low 32 bits == LDS byte offset (ISA §10.2 aperture mapping)
    const u32 lds0 = (u32)(size_t)&smem[0];

    auto issue = [&](int buf, int kb) {
        u32 la = lds0 + (u32)(buf * (BM + BN) * LDT * 2);
        tdm_load_2d(la,                    A  + (size_t)m0 * K + kb, BM, K);
        tdm_load_2d(la + BM * LDT * 2,     BT + (size_t)n0 * K + kb, BN, K);
    };

    if (wave == 0) { issue(0, 0); __builtin_amdgcn_s_wait_tensorcnt(0); }
    __syncthreads();

    v8f acc[2][4] = {};
    int cur = 0;
    for (int kb = 0; kb < K; kb += BK) {
        if (wave == 0 && kb + BK < K) issue(cur ^ 1, kb + BK);

        const __bf16* Abuf = smem + cur * (BM + BN) * LDT;
        const __bf16* Bbuf = Abuf + BM * LDT;

        Frag16 af[2];
        #pragma unroll
        for (int s = 0; s < 2; ++s) {
            const u32* row = (const u32*)(Abuf + (wave * 32 + s * 16 + lr) * LDT);
            #pragma unroll
            for (int d = 0; d < 4; ++d) {
                af[s].u[d]     = row[half * 4 + d];
                af[s].u[4 + d] = row[half * 4 + 8 + d];
            }
        }
        Frag16 bfr[4];
        #pragma unroll
        for (int t = 0; t < 4; ++t) {
            const u32* row = (const u32*)(Bbuf + (t * 16 + lr) * LDT);
            #pragma unroll
            for (int d = 0; d < 4; ++d) { bfr[t].u[d] = row[half * 4 + d]; bfr[t].u[4 + d] = row[half * 4 + 8 + d]; }
        }
        #pragma unroll
        for (int s = 0; s < 2; ++s)
            #pragma unroll
            for (int t = 0; t < 4; ++t)
                acc[s][t] = __builtin_amdgcn_wmma_f32_16x16x32_bf16(
                    false, af[s].v, false, bfr[t].v, (short)0, acc[s][t], false, false);

        __syncthreads();
        if (wave == 0 && kb + BK < K) __builtin_amdgcn_s_wait_tensorcnt(0);
        __syncthreads();
        cur ^= 1;
    }

    float* cb = C + (size_t)(m0 + wave * 32 + half * 8) * N + n0 + lr;
    #pragma unroll
    for (int s = 0; s < 2; ++s)
        #pragma unroll
        for (int t = 0; t < 4; ++t)
            #pragma unroll
            for (int r = 0; r < 8; ++r)
                cb[(size_t)(s * 16 + r) * N + t * 16] = acc[s][t][r];
}

// =====================================================================
// f32 -> bf16 convert (vectorized x4)
// =====================================================================
__global__ __launch_bounds__(256)
void convert_bf16(const float* __restrict__ in, __bf16* __restrict__ out, size_t n4)
{
    size_t i = (size_t)blockIdx.x * 256 + threadIdx.x;
    if (i < n4) {
        float4 v = ((const float4*)in)[i];
        v4bf o = { (__bf16)v.x, (__bf16)v.y, (__bf16)v.z, (__bf16)v.w };
        ((v4bf*)out)[i] = o;
    }
}

// =====================================================================
// Transpose + convert: in[rows][cols] f32 -> out[cols][rows] bf16
// =====================================================================
__global__ __launch_bounds__(256)
void transpose_bf16(const float* __restrict__ in, __bf16* __restrict__ out,
                    int rows, int cols)
{
    __shared__ float tile[32][33];
    const int bx = blockIdx.x * 32;
    const int by = blockIdx.y * 32;
    const int tx = threadIdx.x & 31, ty = threadIdx.x >> 5;
    #pragma unroll
    for (int i = 0; i < 32; i += 8)
        tile[ty + i][tx] = in[(size_t)(by + ty + i) * cols + bx + tx];
    __syncthreads();
    #pragma unroll
    for (int i = 0; i < 32; i += 8)
        out[(size_t)(bx + ty + i) * rows + by + tx] = (__bf16)tile[tx][ty + i];
}

// =====================================================================
// RMSNorm -> bf16
// =====================================================================
__global__ __launch_bounds__(256)
void rmsnorm_kernel(const float* __restrict__ x, const float* __restrict__ w,
                    __bf16* __restrict__ y, int width, int in_stride)
{
    const int row = blockIdx.x;
    const float* xr = x + (size_t)row * in_stride;
    __shared__ float red[256];
    float s = 0.f;
    for (int c = threadIdx.x; c < width; c += 256) { float v = xr[c]; s += v * v; }
    red[threadIdx.x] = s;
    __syncthreads();
    for (int k = 128; k > 0; k >>= 1) {
        if (threadIdx.x < k) red[threadIdx.x] += red[threadIdx.x + k];
        __syncthreads();
    }
    const float r = rsqrtf(red[0] / (float)width + EPS);
    for (int c = threadIdx.x; c < width; c += 256)
        y[(size_t)row * width + c] = (__bf16)(xr[c] * r * w[c]);
}

// =====================================================================
// Build bf16 Q / K (RoPE'd) and transposed V
// =====================================================================
__global__ __launch_bounds__(128)
void build_qkv(const int* __restrict__ positions,
               const float* __restrict__ q,       // T x 3072
               const float* __restrict__ kv,      // T x 4096
               const float* __restrict__ latent,  // T x 576
               __bf16* __restrict__ Qb, __bf16* __restrict__ Kb,
               __bf16* __restrict__ Vt)
{
    const int t = blockIdx.x;
    const int tid = threadIdx.x;
    __shared__ float cs[32], sn[32], kpe[64];

    if (tid < 32) {
        float inv = __expf(-(float)(2 * tid) * (__logf(10000.0f) / 64.0f)) * 0.125f;
        float fr  = (float)positions[t] * inv;
        cs[tid] = __cosf(fr);
        sn[tid] = __sinf(fr);
    }
    __syncthreads();
    if (tid < 32) {
        float x1 = latent[(size_t)t * 576 + KV_LORA + 2 * tid];
        float x2 = latent[(size_t)t * 576 + KV_LORA + 2 * tid + 1];
        kpe[tid]      = x1 * cs[tid] - x2 * sn[tid];
        kpe[32 + tid] = x2 * cs[tid] + x1 * sn[tid];
    }
    __syncthreads();

    for (int idx = tid; idx < N_HEADS * QK_DIM; idx += 128) {
        int h = idx / QK_DIM, d = idx % QK_DIM;
        const float* qrow = q + (size_t)t * (N_HEADS * QK_DIM) + (size_t)h * QK_DIM;
        float val;
        if (d < QK_NOPE) {
            val = qrow[d];
        } else {
            int j = (d - QK_NOPE) & 31;
            bool hi = (d - QK_NOPE) >= 32;
            float x1 = qrow[QK_NOPE + 2 * j];
            float x2 = qrow[QK_NOPE + 2 * j + 1];
            val = hi ? (x2 * cs[j] + x1 * sn[j]) : (x1 * cs[j] - x2 * sn[j]);
        }
        Qb[(size_t)t * (N_HEADS * QK_DIM) + idx] = (__bf16)val;
    }
    for (int idx = tid; idx < N_HEADS * QK_DIM; idx += 128) {
        int h = idx / QK_DIM, d = idx % QK_DIM;
        float val = (d < QK_NOPE) ? kv[(size_t)t * 4096 + (size_t)h * 256 + d]
                                  : kpe[d - QK_NOPE];
        Kb[(size_t)t * (N_HEADS * QK_DIM) + idx] = (__bf16)val;
    }
    for (int idx = tid; idx < N_HEADS * V_DIM; idx += 128) {
        int h = idx >> 7, d = idx & 127;
        Vt[((size_t)h * V_DIM + d) * T_TOK + t] =
            (__bf16)kv[(size_t)t * 4096 + (size_t)h * 256 + QK_NOPE + d];
    }
}

// =====================================================================
// Causal flash attention, 1 wave per (head, 16-query block). Outputs bf16.
// K/V fragment loads are depth-2 software-pipelined: the next fragment's
// b128 loads are issued before the WMMAs consuming the current fragment,
// so waitcnt can drain partially (loadcnt<=2) instead of fully.
// =====================================================================
__global__ __launch_bounds__(32)
void flash_attn(const __bf16* __restrict__ Qb, const __bf16* __restrict__ Kb,
                const __bf16* __restrict__ Vt, __bf16* __restrict__ attn)
{
    const int qb = blockIdx.x, h = blockIdx.y;
    const int q0 = qb * 16;
    const int lane = threadIdx.x;
    const int half = lane >> 4;
    const int lr   = lane & 15;

    __shared__ float  s_s[16][32];
    __shared__ __bf16 p_s[16][32];
    __shared__ float  alpha_s[16];

    Frag16 qf[6];
    {
        const u32* qrow = (const u32*)(Qb + ((size_t)(q0 + lr) * N_HEADS + h) * QK_DIM);
        #pragma unroll
        for (int c = 0; c < 6; ++c) {
            int base = c * 16 + half * 8;
            #pragma unroll
            for (int d = 0; d < 4; ++d) {
                qf[c].u[d]     = qrow[base + d];
                qf[c].u[4 + d] = qrow[base + 4 + d];
            }
        }
    }

    v8f o[8] = {};
    float m_i = -INFINITY, l_i = 0.f;

    for (int k0 = 0; k0 < q0 + 16; k0 += 32) {
        const u32* krow0 =
            (const u32*)(Kb + ((size_t)(k0 + lr) * N_HEADS + h) * QK_DIM);
        const u32* krow1 =
            (const u32*)(Kb + ((size_t)(k0 + 16 + lr) * N_HEADS + h) * QK_DIM);

        // ---- S chains with depth-2 pipelined K-fragment loads ----
        Frag16 k0f, k1f;
        #pragma unroll
        for (int d = 0; d < 8; ++d) k0f.u[d] = krow0[half * 8 + d];
        #pragma unroll
        for (int d = 0; d < 8; ++d) k1f.u[d] = krow1[half * 8 + d];

        v8f s0 = {}, s1 = {};
        #pragma unroll
        for (int c = 0; c < 6; ++c) {
            Frag16 n0, n1;
            if (c < 5) {
                int base = (c + 1) * 16 + half * 8;
                #pragma unroll
                for (int d = 0; d < 8; ++d) n0.u[d] = krow0[base + d];
                #pragma unroll
                for (int d = 0; d < 8; ++d) n1.u[d] = krow1[base + d];
            }
            s0 = __builtin_amdgcn_wmma_f32_16x16x32_bf16(
                     false, qf[c].v, false, k0f.v, (short)0, s0, false, false);
            s1 = __builtin_amdgcn_wmma_f32_16x16x32_bf16(
                     false, qf[c].v, false, k1f.v, (short)0, s1, false, false);
            if (c < 5) { k0f = n0; k1f = n1; }
        }
        #pragma unroll
        for (int r = 0; r < 8; ++r) {
            int rowm = r + half * 8;
            float v0 = s0[r] * SCALING;
            float v1 = s1[r] * SCALING;
            if (k0 + lr      > q0 + rowm) v0 = -INFINITY;
            if (k0 + 16 + lr > q0 + rowm) v1 = -INFINITY;
            s_s[rowm][lr]      = v0;
            s_s[rowm][16 + lr] = v1;
        }
        __syncthreads();

        // ---- online softmax (one row per lane, lanes 0..15) ----
        if (lane < 16) {
            const int row = lane;
            float mx = m_i;
            #pragma unroll
            for (int c = 0; c < 32; ++c) mx = fmaxf(mx, s_s[row][c]);
            float alpha = __expf(m_i - mx);
            float sum = 0.f;
            #pragma unroll
            for (int c = 0; c < 32; ++c) {
                float p = __expf(s_s[row][c] - mx);
                sum += p;
                p_s[row][c] = (__bf16)p;
            }
            l_i = l_i * alpha + sum;
            m_i = mx;
            alpha_s[row] = alpha;
        }
        __syncthreads();

        // ---- rescale O accumulators ----
        #pragma unroll
        for (int t = 0; t < 8; ++t)
            #pragma unroll
            for (int r = 0; r < 8; ++r)
                o[t][r] *= alpha_s[r + half * 8];

        // ---- P fragment (A layout) from LDS ----
        Frag16 pf;
        {
            const u32* prow = (const u32*)&p_s[lr][0];
            #pragma unroll
            for (int d = 0; d < 4; ++d) {
                pf.u[d]     = prow[half * 8 + d];
                pf.u[4 + d] = prow[half * 8 + 4 + d];
            }
        }
        // ---- PV with depth-2 pipelined V-fragment loads ----
        const int vbase = (k0 >> 1) + half * 8;
        Frag16 vfc;
        {
            const u32* vrow = (const u32*)(Vt + ((size_t)h * V_DIM + lr) * T_TOK);
            #pragma unroll
            for (int d = 0; d < 8; ++d) vfc.u[d] = vrow[vbase + d];
        }
        #pragma unroll
        for (int t = 0; t < 8; ++t) {
            Frag16 vn;
            if (t < 7) {
                const u32* vrow =
                    (const u32*)(Vt + ((size_t)h * V_DIM + (t + 1) * 16 + lr) * T_TOK);
                #pragma unroll
                for (int d = 0; d < 8; ++d) vn.u[d] = vrow[vbase + d];
            }
            o[t] = __builtin_amdgcn_wmma_f32_16x16x32_bf16(
                       false, pf.v, false, vfc.v, (short)0, o[t], false, false);
            if (t < 7) vfc = vn;
        }
        __syncthreads();
    }

    if (lane < 16) alpha_s[lane] = 1.0f / l_i;
    __syncthreads();
    #pragma unroll
    for (int t = 0; t < 8; ++t)
        #pragma unroll
        for (int r = 0; r < 8; ++r) {
            int rowm = r + half * 8;
            attn[((size_t)(q0 + rowm) * N_HEADS + h) * V_DIM + t * 16 + lr] =
                (__bf16)(o[t][r] * alpha_s[rowm]);
        }
}

// =====================================================================
// Host-side orchestration
// =====================================================================
extern "C" void kernel_launch(void* const* d_in, const int* in_sizes, int n_in,
                              void* d_out, int out_size, void* d_ws, size_t ws_size,
                              hipStream_t stream)
{
    (void)in_sizes; (void)n_in; (void)out_size; (void)ws_size;

    const int*   positions   = (const int*)  d_in[0];
    const float* hidden      = (const float*)d_in[1];
    const float* Wqa         = (const float*)d_in[2];
    const float* q_a_norm_w  = (const float*)d_in[3];
    const float* Wqb         = (const float*)d_in[4];
    const float* Wkva        = (const float*)d_in[5];
    const float* kv_a_norm_w = (const float*)d_in[6];
    const float* Wkvb        = (const float*)d_in[7];
    const float* Wo          = (const float*)d_in[8];
    float* out = (float*)d_out;

    char* ws = (char*)d_ws;
    size_t off = 0;
    auto alloc = [&](size_t bytes) { char* p = ws + off; off += (bytes + 255) & ~size_t(255); return p; };

    __bf16* hidden_bf = (__bf16*)alloc((size_t)T_TOK * HIDDEN * 2);
    __bf16* WqaT   = (__bf16*)alloc((size_t)Q_LORA * HIDDEN * 2);
    __bf16* WkvaT  = (__bf16*)alloc((size_t)(KV_LORA + QK_ROPE) * HIDDEN * 2);
    __bf16* WqbT   = (__bf16*)alloc((size_t)(N_HEADS * QK_DIM) * Q_LORA * 2);
    __bf16* WkvbT  = (__bf16*)alloc((size_t)(N_HEADS * 256) * KV_LORA * 2);
    __bf16* WoT    = (__bf16*)alloc((size_t)HIDDEN * HIDDEN * 2);
    float*  q_a    = (float*) alloc((size_t)T_TOK * Q_LORA * 4);
    __bf16* q_a_bf = (__bf16*)alloc((size_t)T_TOK * Q_LORA * 2);
    float*  latent = (float*) alloc((size_t)T_TOK * (KV_LORA + QK_ROPE) * 4);
    __bf16* kv_a_bf= (__bf16*)alloc((size_t)T_TOK * KV_LORA * 2);
    float*  qfull  = (float*) alloc((size_t)T_TOK * N_HEADS * QK_DIM * 4);
    float*  kvfull = (float*) alloc((size_t)T_TOK * N_HEADS * 256 * 4);
    __bf16* Qb     = (__bf16*)alloc((size_t)T_TOK * N_HEADS * QK_DIM * 2);
    __bf16* Kb     = (__bf16*)alloc((size_t)T_TOK * N_HEADS * QK_DIM * 2);
    __bf16* Vt     = (__bf16*)alloc((size_t)N_HEADS * V_DIM * T_TOK * 2);
    __bf16* attn_bf= (__bf16*)alloc((size_t)T_TOK * N_HEADS * V_DIM * 2);

    {
        size_t n4 = (size_t)T_TOK * HIDDEN / 4;
        convert_bf16<<<(unsigned)((n4 + 255) / 256), 256, 0, stream>>>(hidden, hidden_bf, n4);
    }
    transpose_bf16<<<dim3(Q_LORA / 32, HIDDEN / 32), 256, 0, stream>>>(Wqa, WqaT, HIDDEN, Q_LORA);
    transpose_bf16<<<dim3((KV_LORA + QK_ROPE) / 32, HIDDEN / 32), 256, 0, stream>>>(Wkva, WkvaT, HIDDEN, KV_LORA + QK_ROPE);
    transpose_bf16<<<dim3((N_HEADS * QK_DIM) / 32, Q_LORA / 32), 256, 0, stream>>>(Wqb, WqbT, Q_LORA, N_HEADS * QK_DIM);
    transpose_bf16<<<dim3((N_HEADS * 256) / 32, KV_LORA / 32), 256, 0, stream>>>(Wkvb, WkvbT, KV_LORA, N_HEADS * 256);
    transpose_bf16<<<dim3(HIDDEN / 32, HIDDEN / 32), 256, 0, stream>>>(Wo, WoT, HIDDEN, HIDDEN);

    dim3 blk(128);
    gemm_tdm<Q_LORA, HIDDEN><<<dim3(T_TOK / BM, Q_LORA / BN), blk, 0, stream>>>(hidden_bf, WqaT, q_a);
    gemm_tdm<KV_LORA + QK_ROPE, HIDDEN><<<dim3(T_TOK / BM, (KV_LORA + QK_ROPE) / BN), blk, 0, stream>>>(hidden_bf, WkvaT, latent);
    rmsnorm_kernel<<<T_TOK, 256, 0, stream>>>(q_a, q_a_norm_w, q_a_bf, Q_LORA, Q_LORA);
    rmsnorm_kernel<<<T_TOK, 256, 0, stream>>>(latent, kv_a_norm_w, kv_a_bf, KV_LORA, KV_LORA + QK_ROPE);
    gemm_tdm<N_HEADS * QK_DIM, Q_LORA><<<dim3(T_TOK / BM, (N_HEADS * QK_DIM) / BN), blk, 0, stream>>>(q_a_bf, WqbT, qfull);
    gemm_tdm<N_HEADS * 256, KV_LORA><<<dim3(T_TOK / BM, (N_HEADS * 256) / BN), blk, 0, stream>>>(kv_a_bf, WkvbT, kvfull);
    build_qkv<<<T_TOK, 128, 0, stream>>>(positions, qfull, kvfull, latent, Qb, Kb, Vt);
    flash_attn<<<dim3(T_TOK / 16, N_HEADS), 32, 0, stream>>>(Qb, Kb, Vt, attn_bf);
    gemm_tdm<HIDDEN, HIDDEN><<<dim3(T_TOK / BM, HIDDEN / BN), blk, 0, stream>>>(attn_bf, WoT, out);
}
